// ClassicalSelfAttention_65481071406032
// MI455X (gfx1250) — matrix operand order repaired and checked by reading, not compiled, not run on hardware
//
#include <hip/hip_runtime.h>
#include <hip/hip_bf16.h>
#include <stdint.h>

typedef __bf16 bf16_t;
typedef __attribute__((ext_vector_type(16))) __bf16 v16bf;
typedef __attribute__((ext_vector_type(8)))  __bf16 v8bf;
typedef __attribute__((ext_vector_type(8)))  float  v8f;

#define EMBED 1024
#define SEQ   4096
#define BATCH 4
#define ROWS  (BATCH * SEQ)   // 16384

// Flip to 0 if the assembler rejects global_load_async_to_lds_b128.
#define USE_ASYNC_LDS 1

static __device__ __forceinline__ bf16_t f2bf(float f) {
  unsigned u = __builtin_bit_cast(unsigned, f);
  unsigned r = (u + 0x7fffu + ((u >> 16) & 1u)) >> 16;   // round-to-nearest-even
  unsigned short s = (unsigned short)r;
  return __builtin_bit_cast(bf16_t, s);
}
static __device__ __forceinline__ float bf2f(bf16_t b) {
  unsigned short s = __builtin_bit_cast(unsigned short, b);
  unsigned u = ((unsigned)s) << 16;
  return __builtin_bit_cast(float, u);
}

// ---------------- weight prep: W^T in bf16, and column-mean of Wv ------------
__global__ void __launch_bounds__(256) prep_wt(const float* __restrict__ Wq,
                                               const float* __restrict__ Wk,
                                               bf16_t* __restrict__ WqT,
                                               bf16_t* __restrict__ WkT) {
  int gid = blockIdx.x * 256 + threadIdx.x;      // over 1M elements of W^T
  int n = gid >> 10;                             // output column of W
  int k = gid & 1023;                            // input dim
  WqT[gid] = f2bf(Wq[(size_t)k * EMBED + n]);
  WkT[gid] = f2bf(Wk[(size_t)k * EMBED + n]);
}

__global__ void __launch_bounds__(256) prep_wvbar(const float* __restrict__ Wv,
                                                  float* __restrict__ wvbar) {
  int k = blockIdx.x * 256 + threadIdx.x;        // 0..1023
  const float* row = Wv + (size_t)k * EMBED;
  float s = 0.f;
  for (int n = 0; n < EMBED; ++n) s += row[n];
  wvbar[k] = s * (1.0f / EMBED);
}

// ---------------- Q/K projection (bf16 WMMA, f32 acc) + vbar -----------------
#define XPITCH 1040   // bf16 elems per LDS row (16-elem pad; 16B-aligned chunks)

__global__ void __launch_bounds__(256) qk_proj(const float* __restrict__ X,
                                               const bf16_t* __restrict__ WqT,
                                               const bf16_t* __restrict__ WkT,
                                               const float* __restrict__ wvbar,
                                               bf16_t* __restrict__ Qb,
                                               bf16_t* __restrict__ Kb,
                                               float* __restrict__ vbar) {
  __shared__ bf16_t xs[16 * XPITCH];
  const int tid = threadIdx.x;
  const int m0  = blockIdx.x * 16;               // global row base

  // Stage X tile (16 x 1024 f32) into LDS as bf16.
  {
    const float4* src = (const float4*)(X + (size_t)m0 * EMBED); // 16*256 float4
    for (int i = tid; i < 16 * 256; i += 256) {
      float4 v = src[i];
      int row = i >> 8;
      int col = (i & 255) << 2;
      bf16_t* d = &xs[row * XPITCH + col];
      d[0] = f2bf(v.x); d[1] = f2bf(v.y); d[2] = f2bf(v.z); d[3] = f2bf(v.w);
    }
  }
  __syncthreads();

  // vbar[m] = X[m,:] . wvbar  (collapsed V projection + mean)
  if (tid < 16) {
    float s = 0.f;
    for (int k = 0; k < EMBED; ++k) s += bf2f(xs[tid * XPITCH + k]) * wvbar[k];
    vbar[m0 + tid] = s;
  }

  const int wave = tid >> 5;
  const int lane = tid & 31;
  const int l15  = lane & 15;
  const int hi   = (lane >> 4) & 1;

  const bf16_t* xrow = &xs[l15 * XPITCH + hi * 16];

  for (int t = wave; t < 128; t += 8) {
    const bf16_t* WT  = (t & 64) ? WkT : WqT;
    bf16_t*       Out = (t & 64) ? Kb  : Qb;
    const int n0 = (t & 63) << 4;
    const bf16_t* wcol = WT + (size_t)(n0 + l15) * EMBED + hi * 16;

    v8f acc0 = {}; v8f acc1 = {};
    for (int kc = 0; kc < 32; kc += 2) {
      {
        v8bf alo = *(const v8bf*)(xrow + kc * 32);
        v8bf ahi = *(const v8bf*)(xrow + kc * 32 + 8);
        v16bf a;
        #pragma unroll
        for (int i = 0; i < 8; ++i) { a[i] = alo[i]; a[8 + i] = ahi[i]; }
        v16bf b = *(const v16bf*)(wcol + kc * 32);
        acc0 = __builtin_amdgcn_wmma_f32_16x16x32_bf16(false, a, false, b,
                                                       (short)0, acc0, false, false);
      }
      {
        v8bf alo = *(const v8bf*)(xrow + kc * 32 + 32);
        v8bf ahi = *(const v8bf*)(xrow + kc * 32 + 40);
        v16bf a;
        #pragma unroll
        for (int i = 0; i < 8; ++i) { a[i] = alo[i]; a[8 + i] = ahi[i]; }
        v16bf b = *(const v16bf*)(wcol + (kc + 1) * 32);
        acc1 = __builtin_amdgcn_wmma_f32_16x16x32_bf16(false, a, false, b,
                                                       (short)0, acc1, false, false);
      }
    }
    #pragma unroll
    for (int j = 0; j < 8; ++j) {
      float r = acc0[j] + acc1[j];
      Out[(size_t)(m0 + hi * 8 + j) * EMBED + n0 + l15] = f2bf(r);
    }
  }
}

// ---------------- K-tile staging (global -> LDS) -----------------------------
// One 16x1024 bf16 tile = 2048 16B chunks; 256 threads x 8 chunks each.
static __device__ __forceinline__ void stage_k(bf16_t* dst_lds,
                                               const bf16_t* __restrict__ src_g,
                                               int tid) {
#if USE_ASYNC_LDS
  #pragma unroll
  for (int s = 0; s < 8; ++s) {
    int idx = s * 256 + tid;
    int row = idx >> 7;          // 128 chunks per row
    int col = idx & 127;
    unsigned loff = (unsigned)(uintptr_t)(dst_lds + row * XPITCH + col * 8);
    const bf16_t* g = src_g + row * EMBED + col * 8;
    asm volatile("global_load_async_to_lds_b128 %0, %1, off"
                 :: "v"(loff), "v"(g) : "memory");
  }
#else
  #pragma unroll
  for (int s = 0; s < 8; ++s) {
    int idx = s * 256 + tid;
    int row = idx >> 7;
    int col = idx & 127;
    v8bf v = *(const v8bf*)(src_g + row * EMBED + col * 8);
    *(v8bf*)(dst_lds + row * XPITCH + col * 8) = v;
  }
#endif
}

static __device__ __forceinline__ void wait_stage() {
#if USE_ASYNC_LDS
  asm volatile("s_wait_asynccnt 0x0" ::: "memory");
#endif
}

// ---------------- flash-style attention with collapsed V ---------------------
// Block = 8 waves = 128 query rows (one batch). K tiles staged once per block
// into double-buffered LDS (shared by all 8 waves); each wave's Q fragments
// are hoisted into 256 VGPRs and reused across all 256 key tiles.
__global__ void __launch_bounds__(256, 1) attn(const bf16_t* __restrict__ Qb,
                                               const bf16_t* __restrict__ Kb,
                                               const float* __restrict__ vbar,
                                               float* __restrict__ out) {
  __shared__ bf16_t ks[2 * 16 * XPITCH];         // double-buffered K tile

  const int tid  = threadIdx.x;
  const int wave = tid >> 5;
  const int lane = tid & 31;
  const int l15  = lane & 15;
  const int hi   = (lane >> 4) & 1;

  const int qt    = blockIdx.x * 8 + wave;       // 0..1023
  const int m0    = qt * 16;                     // global query row base
  const int kbase = (m0 >> 12) << 12;            // key row base of this batch

  // Hoist Q fragments (B operand: lane col = query l15, K = hi*16..+15).
  const bf16_t* qrow = Qb + (size_t)(m0 + l15) * EMBED + hi * 16;
  v16bf qf[32];
  #pragma unroll
  for (int kc = 0; kc < 32; ++kc) qf[kc] = *(const v16bf*)(qrow + kc * 32);

  float m_run = -1e30f, ssum = 0.f, vacc = 0.f;
  const float scale = 0.03125f;                  // 1/sqrt(1024)

  // Prologue: stage tile 0.
  stage_k(ks, Kb + (size_t)kbase * EMBED, tid);

  for (int kt = 0; kt < 256; ++kt) {
    wait_stage();                                // stage(kt) complete (this thread)
    __syncthreads();                             // publish buf[kt&1] to all waves
    if (kt + 1 < 256)                            // issue next tile AFTER barrier:
      stage_k(ks + ((kt + 1) & 1) * (16 * XPITCH),
              Kb + (size_t)(kbase + (kt + 1) * 16) * EMBED, tid);

    const int krow0 = kbase + kt * 16;
    const bf16_t* abase = ks + (kt & 1) * (16 * XPITCH) + l15 * XPITCH + hi * 16;

    v8f acc0 = {}, acc1 = {};
    #pragma unroll
    for (int kc = 0; kc < 32; kc += 2) {
      {
        v8bf alo = *(const v8bf*)(abase + kc * 32);
        v8bf ahi = *(const v8bf*)(abase + kc * 32 + 8);
        v16bf a;
        #pragma unroll
        for (int i = 0; i < 8; ++i) { a[i] = alo[i]; a[8 + i] = ahi[i]; }
        acc0 = __builtin_amdgcn_wmma_f32_16x16x32_bf16(false, a, false, qf[kc],
                                                       (short)0, acc0, false, false);
      }
      {
        v8bf alo = *(const v8bf*)(abase + kc * 32 + 32);
        v8bf ahi = *(const v8bf*)(abase + kc * 32 + 40);
        v16bf a;
        #pragma unroll
        for (int i = 0; i < 8; ++i) { a[i] = alo[i]; a[8 + i] = ahi[i]; }
        acc1 = __builtin_amdgcn_wmma_f32_16x16x32_bf16(false, a, false, qf[kc + 1],
                                                       (short)0, acc1, false, false);
      }
    }

    // Lane holds S^T[key = krow0+hi*8+j, query = l15] in acc[j].
    float s[8];
    float tmax = -1e30f;
    #pragma unroll
    for (int j = 0; j < 8; ++j) {
      s[j] = (acc0[j] + acc1[j]) * scale;
      tmax = fmaxf(tmax, s[j]);
    }
    float omax = __shfl_xor(tmax, 16);
    float mnew = fmaxf(m_run, fmaxf(tmax, omax));
    float alpha = __expf(m_run - mnew);

    const float* vb = vbar + krow0 + hi * 8;
    float psum = 0.f, pv = 0.f;
    #pragma unroll
    for (int j = 0; j < 8; ++j) {
      float p = __expf(s[j] - mnew);
      psum += p;
      pv   += p * vb[j];
    }
    ssum  = ssum * alpha + psum;
    vacc  = vacc * alpha + pv;
    m_run = mnew;
  }

  float tot_s = ssum + __shfl_xor(ssum, 16);
  float tot_v = vacc + __shfl_xor(vacc, 16);
  if (hi == 0) out[m0 + l15] = tot_v / tot_s;
}

// ---------------- launch -----------------------------------------------------
extern "C" void kernel_launch(void* const* d_in, const int* in_sizes, int n_in,
                              void* d_out, int out_size, void* d_ws, size_t ws_size,
                              hipStream_t stream) {
  (void)in_sizes; (void)n_in; (void)out_size; (void)ws_size;
  const float* X  = (const float*)d_in[0];
  const float* Wq = (const float*)d_in[1];
  const float* Wk = (const float*)d_in[2];
  const float* Wv = (const float*)d_in[3];

  char* ws = (char*)d_ws;
  // layout: WqT(2MB) | WkT(2MB) | wvbar(4KB) | Qb(32MB) | Kb(32MB) | vbar(64KB)
  bf16_t* WqT  = (bf16_t*)(ws);
  bf16_t* WkT  = (bf16_t*)(ws + (size_t)(2u << 20));
  float*  wvb  = (float*) (ws + (size_t)(4u << 20));
  bf16_t* Qb   = (bf16_t*)(ws + (size_t)(4u << 20) + 4096);
  bf16_t* Kb   = (bf16_t*)(ws + (size_t)(4u << 20) + 4096 + (size_t)(32u << 20));
  float*  vbar = (float*) (ws + (size_t)(4u << 20) + 4096 + (size_t)(64u << 20));
  float*  out  = (float*)d_out;

  prep_wt   <<<4096, 256, 0, stream>>>(Wq, Wk, WqT, WkT);
  prep_wvbar<<<   4, 256, 0, stream>>>(Wv, wvb);
  qk_proj   <<<1024, 256, 0, stream>>>(X, WqT, WkT, wvb, Qb, Kb, vbar);
  attn      <<< 128, 256, 0, stream>>>(Qb, Kb, vbar, out);
}
